// BlockWiseHistogramEncoder_42056319762524
// MI455X (gfx1250) — compile-verified, matrix operand order checked by reading, not run on hardware
//
#include <hip/hip_runtime.h>

// Block-wise label histogram via IU8 WMMA:
//   per 16x16 block: D(16x16 i32) = ones(16x64 u8) x ByteCounters(64x16 u8)
// Byte counters are built with packed ds_add_u32 (4 elements per K-row -> max 4
// per byte, no overflow), stored class-major in LDS so the WMMA B operand is a
// contiguous 2x ds_load_b128 per lane.

typedef int v8i __attribute__((ext_vector_type(8)));

#define IMG 512
#define WAVES_PER_WG 4

__global__ __launch_bounds__(128) void
BlockWiseHistogramEncoder_kernel(const int* __restrict__ x, float* __restrict__ out) {
    const unsigned tid  = threadIdx.x;
    const unsigned lane = tid & 31u;
    const unsigned wave = tid >> 5;                          // 0..3
    const unsigned gblk = blockIdx.x * WAVES_PER_WG + wave;  // 0..16383
    const unsigned b    = gblk >> 10;                        // batch
    const unsigned l    = gblk & 1023u;                      // block index in image
    const unsigned lr   = l >> 5, lc = l & 31u;              // block row/col (32x32 grid)

    // 64 K-rows x 64 class-bytes per wave = 1024 dwords = 4KB
    __shared__ unsigned lds[WAVES_PER_WG * 1024];
    const unsigned slice = wave * 1024u;                     // dword offset of this wave's slice

    // ---- zero counters: 32 dwords/lane as 8x uint4 stores ----
    {
        uint4 z = make_uint4(0u, 0u, 0u, 0u);
        uint4* p = reinterpret_cast<uint4*>(&lds[slice]);
        #pragma unroll
        for (int j = 0; j < 8; ++j) p[lane + 32u * j] = z;
    }
    __syncthreads();

    // ---- load 8 labels per lane (half a block row): 2x global_load_b128 ----
    const unsigned base = b * (IMG * IMG) + (lr * 16u) * IMG + lc * 16u;
    const unsigned addr = base + (lane >> 1) * IMG + (lane & 1u) * 8u;
    const int4 v0 = *reinterpret_cast<const int4*>(x + addr);
    const int4 v1 = *reinterpret_cast<const int4*>(x + addr + 4);
    const int lab[8] = {v0.x, v0.y, v0.z, v0.w, v1.x, v1.y, v1.z, v1.w};

    // ---- scatter into packed byte counters: counter byte [class][k], k = lane + 32*(i>=4) ----
    // byte position within dword = lane&3 (constant per lane), dword = class*16 + (k>>2)
    const unsigned inc = 1u << ((lane & 3u) * 8u);
    const unsigned dwl = slice + (lane >> 2);
    #pragma unroll
    for (int i = 0; i < 8; ++i) {
        const unsigned dw = dwl + (unsigned)lab[i] * 16u + (i >= 4 ? 8u : 0u);
        atomicAdd(&lds[dw], inc);   // lowers to non-returning ds_add_u32
    }
    __syncthreads();

    // ---- WMMA reduction: for class group g, D = ones(16x64) * B(64x16) ----
    // A = all-ones bytes => D[m][c] = sum_k B[k][c] for every row m (K-layout permutation
    // invariant). B column c = lane&15 (documented striping); bytes are the class-major
    // counters, loaded contiguously.
    v8i a;
    #pragma unroll
    for (int j = 0; j < 8; ++j) a[j] = 0x01010101;

    const unsigned n      = lane & 15u;
    const unsigned halfdw = (lane >> 4) * 4u;   // K-half per lane pair-of-16

    float vals[4];
    #pragma unroll
    for (int g = 0; g < 4; ++g) {
        const unsigned dw = slice + (16u * g + n) * 16u + halfdw;
        const uint4 lo = *reinterpret_cast<const uint4*>(&lds[dw]);       // ds_load_b128
        const uint4 hi = *reinterpret_cast<const uint4*>(&lds[dw + 8u]);  // ds_load_b128
        v8i bm;
        bm[0] = (int)lo.x; bm[1] = (int)lo.y; bm[2] = (int)lo.z; bm[3] = (int)lo.w;
        bm[4] = (int)hi.x; bm[5] = (int)hi.y; bm[6] = (int)hi.z; bm[7] = (int)hi.w;
        v8i c = {};
        // (sgn_a, A, sgn_b, B, C, reuse_a, reuse_b) -> v_wmma_i32_16x16x64_iu8
        v8i d = __builtin_amdgcn_wmma_i32_16x16x64_iu8(false, a, false, bm, c, false, false);
        vals[g] = (float)d[0] * (1.0f / 256.0f);
    }

    // ---- write 65-bin histogram: bin 0 = 0, bin 1+16g+n = count/256 ----
    float* o = out + (size_t)gblk * 65u;
    if (lane < 16u) {
        #pragma unroll
        for (int g = 0; g < 4; ++g) o[1u + 16u * g + lane] = vals[g];
    }
    if (lane == 0u) o[0] = 0.0f;
}

extern "C" void kernel_launch(void* const* d_in, const int* in_sizes, int n_in,
                              void* d_out, int out_size, void* d_ws, size_t ws_size,
                              hipStream_t stream) {
    const int* x = (const int*)d_in[0];         // (16,1,512,512) int32
    float* out   = (float*)d_out;               // (16,1024,65) f32
    (void)in_sizes; (void)n_in; (void)out_size; (void)d_ws; (void)ws_size;
    dim3 grid(16384 / WAVES_PER_WG);            // 4096 WGs x 4 waves = 16384 blocks
    dim3 block(128);                            // 4 wave32s
    hipLaunchKernelGGL(BlockWiseHistogramEncoder_kernel, grid, block, 0, stream, x, out);
}
